// IPAGNN_41300405518587
// MI455X (gfx1250) — compile-verified, matrix operand order checked by reading, not run on hardware
//
#include <hip/hip_runtime.h>
#include <hip/hip_bf16.h>

typedef __attribute__((ext_vector_type(16))) __bf16 v16bf;
typedef __attribute__((ext_vector_type(8)))  __bf16 v8bf;
typedef __attribute__((ext_vector_type(8)))  float  v8f;

static constexpr int kB = 16;
static constexpr int kN = 512;
static constexpr int kH = 256;
static constexpr int kV = 32000;
static constexpr int kMaxSteps = 48;

// ---------- workspace layout (bytes) ----------
static constexpr size_t SZ_WT     = (size_t)(4 * kH) * kH * 2;        // 1024x256 bf16
static constexpr size_t SZ_NODEBF = (size_t)kB * kN * kH * 2;         // bf16 node emb
static constexpr size_t SZ_XZ     = (size_t)kB * kN * (4 * kH) * 4;   // f32 x@Wx+b
static constexpr size_t SZ_STATE  = (size_t)kB * kN * kH * 4;         // f32 [B,N,H]
static constexpr size_t SZ_IP     = (size_t)kB * kN * 4;

static constexpr size_t OFF_WHT    = 0;
static constexpr size_t OFF_WXT    = OFF_WHT    + SZ_WT;
static constexpr size_t OFF_NODEBF = OFF_WXT    + SZ_WT;
static constexpr size_t OFF_XZ     = OFF_NODEBF + SZ_NODEBF;
static constexpr size_t OFF_C      = OFF_XZ     + SZ_XZ;
static constexpr size_t OFF_H      = OFF_C      + SZ_STATE;
static constexpr size_t OFF_C2     = OFF_H      + SZ_STATE;
static constexpr size_t OFF_H2     = OFF_C2     + SZ_STATE;
static constexpr size_t OFF_IP     = OFF_H2     + SZ_STATE;
static constexpr size_t OFF_IPACC  = OFF_IP     + SZ_IP;
static constexpr size_t OFF_CACC   = OFF_IPACC  + SZ_IP;
static constexpr size_t OFF_HACC   = OFF_CACC   + SZ_STATE;

// ---------- LDS layout for the main kernel (dynamic, 320 KB = WGP max) ----
static constexpr unsigned LDS_H_BYTES    = (unsigned)kN * kH * 2;       // 256 KB bf16 h
static constexpr unsigned LDS_WBUF_BYTE  = LDS_H_BYTES;                 // double-buffered W chunk
static constexpr unsigned LDS_WCHUNK_B   = 4 * 16 * kH * 2;             // 32 KB (4 gates x 16 rows x 256 K)
static constexpr unsigned LDS_TOTAL      = LDS_WBUF_BYTE + 2 * LDS_WCHUNK_B;  // 327680

__device__ __forceinline__ float sigmoidf_(float x) {
    return 1.0f / (1.0f + __expf(-x));
}

// Load one 16x32 bf16 WMMA fragment.  Row-major source (global or LDS) with
// leading dim `ld` (bf16 elements).  Lanes 0-15: row = base+lane,
// K = kbase+0..7 / +16..23.  Lanes 16-31: same rows, K = +8..15 / +24..31.
__device__ __forceinline__ v16bf load_bf16_frag(const __bf16* base,
                                                int row, int kbase, int ld) {
    const int L = threadIdx.x & 31;
    const int r = row + (L & 15);
    const int k0 = kbase + ((L >> 4) << 3);
    const __bf16* p = base + (size_t)r * ld + k0;
    v8bf lo = *reinterpret_cast<const v8bf*>(p);
    v8bf hi = *reinterpret_cast<const v8bf*>(p + 16);
    v16bf out;
#pragma unroll
    for (int i = 0; i < 8; ++i) { out[i] = lo[i]; out[i + 8] = hi[i]; }
    return out;
}

#define WMMA_BF16(A, Bm, C) \
    __builtin_amdgcn_wmma_f32_16x16x32_bf16(false, (A), false, (Bm), (short)0, (C), false, false)

// Async 16-byte global -> LDS copy (per lane), tracked by ASYNCcnt.
__device__ __forceinline__ void async_copy16(unsigned lds_byte, const void* gptr) {
    asm volatile("global_load_async_to_lds_b128 %0, %1, off"
                 :: "v"(lds_byte), "v"(gptr)
                 : "memory");
}
__device__ __forceinline__ void wait_async0() {
    asm volatile("s_wait_asynccnt 0" ::: "memory");
}

// Stage the 4-gate x 16-row x 256-K WhT chunk for hidden-tile `ht` into LDS
// buffer `buf`.  64 rows x 512 B; each wave issues 2 rows, each lane 16 B.
__device__ __forceinline__ void issue_wchunk(const __bf16* __restrict__ WhT,
                                             int ht, int buf, int wave, int lane) {
#pragma unroll
    for (int rr = 0; rr < 2; ++rr) {
        const int rowid = wave * 2 + rr;            // 0..63
        const int g = rowid >> 4, r = rowid & 15;
        const char* gsrc = (const char*)(WhT + (size_t)(g * kH + ht * 16 + r) * kH)
                           + lane * 16;
        const unsigned ldsb = LDS_WBUF_BYTE + (unsigned)buf * LDS_WCHUNK_B
                              + (unsigned)(rowid * 512 + lane * 16);
        async_copy16(ldsb, gsrc);
    }
}

// ---------------------------------------------------------------------------
// Kernel 1: bf16 cast of node embeddings; bf16 transposed weights
// ---------------------------------------------------------------------------
__global__ void ipagnn_prep(const float* __restrict__ Wx, const float* __restrict__ Wh,
                            const float* __restrict__ node,
                            __bf16* __restrict__ WxT, __bf16* __restrict__ WhT,
                            __bf16* __restrict__ nodebf) {
    const int tid = blockIdx.x * blockDim.x + threadIdx.x;
    const int stride = gridDim.x * blockDim.x;
    for (int i = tid; i < 4 * kH * kH; i += stride) {
        int j = i / kH;          // column of W (0..1023)
        int k = i % kH;          // row of W    (0..255)
        WxT[i] = (__bf16)Wx[(size_t)k * (4 * kH) + j];
        WhT[i] = (__bf16)Wh[(size_t)k * (4 * kH) + j];
    }
    for (int i = tid; i < kB * kN * kH; i += stride)
        nodebf[i] = (__bf16)node[i];
}

// ---------------------------------------------------------------------------
// Kernel 2: Xz = node_emb @ Wx + b  (step-invariant).  One wave per 16x16 tile.
// ---------------------------------------------------------------------------
__global__ __launch_bounds__(256)
void ipagnn_xz(const __bf16* __restrict__ nodebf, const __bf16* __restrict__ WxT,
               const float* __restrict__ bvec, float* __restrict__ Xz) {
    const int lane = threadIdx.x & 31;
    const int wave = threadIdx.x >> 5;
    const int tile = blockIdx.x * 8 + wave;   // 0 .. 32767
    const int mt = tile & 511;                // 512 row tiles over B*N rows
    const int nt = tile >> 9;                 // 64 col tiles over 4H cols
    v8f acc = {0.f, 0.f, 0.f, 0.f, 0.f, 0.f, 0.f, 0.f};
#pragma unroll
    for (int kt = 0; kt < 8; ++kt) {
        v16bf a = load_bf16_frag(nodebf, mt * 16, kt * 32, kH);
        v16bf bm = load_bf16_frag(WxT, nt * 16, kt * 32, kH);
        acc = WMMA_BF16(a, bm, acc);
    }
    const int col = nt * 16 + (lane & 15);
    const float bias = bvec[col];
#pragma unroll
    for (int r = 0; r < 8; ++r) {
        int row = mt * 16 + r + ((lane >> 4) << 3);
        Xz[(size_t)row * (4 * kH) + col] = acc[r] + bias;
    }
}

// ---------------------------------------------------------------------------
// Kernel 3: 48-step recurrence.  One 32-wave workgroup per example.
// LDS: h (bf16, 256 KB) + double-buffered WhT chunks (2 x 32 KB, async-filled).
// ---------------------------------------------------------------------------
__global__ __launch_bounds__(1024)
void ipagnn_main(const float* __restrict__ Xz, const __bf16* __restrict__ WhT,
                 float* c, float* h, float* c2b, float* h2b,
                 float* ip, float* ipacc, float* cacc, float* hacc,
                 const float* __restrict__ Wb, const float* __restrict__ bb,
                 const int* __restrict__ tIdxAll, const int* __restrict__ fIdxAll,
                 const int* __restrict__ exitIdx, const int* __restrict__ stepsArr) {
    extern __shared__ char smem[];
    __bf16* hlds = (__bf16*)smem;                       // [512][256] bf16
    __bf16* wbuf = (__bf16*)(smem + LDS_WBUF_BYTE);     // 2 x [64][256] bf16

    const int bEx = blockIdx.x;
    const int tid = threadIdx.x;
    const int lane = tid & 31;
    const int lane15 = lane & 15;
    const int wave = tid >> 5;

    const size_t exOff = (size_t)bEx * kN * kH;
    c += exOff; h += exOff; c2b += exOff; h2b += exOff;
    cacc += exOff; hacc += exOff;
    ip += (size_t)bEx * kN; ipacc += (size_t)bEx * kN;
    const float* XzE = Xz + (size_t)bEx * kN * (4 * kH);
    const int* tI = tIdxAll + bEx * kN;
    const int* fI = fIdxAll + bEx * kN;
    const int exitn = exitIdx[bEx];
    int nsteps = stepsArr[bEx];
    if (nsteps > kMaxSteps) nsteps = kMaxSteps;

    // ---- init: c=h=0 (h mirrored in LDS), ip = one_hot(0) ----
    for (int i = tid; i < kN * kH; i += 1024) {
        c[i] = 0.f; h[i] = 0.f; hlds[i] = (__bf16)0.f;
    }
    for (int n = tid; n < kN; n += 1024) ip[n] = (n == 0) ? 1.f : 0.f;
    __syncthreads();

    for (int it = 0; it < nsteps; ++it) {
        // ---- phase 0: zero segment-sum accumulators ----
        for (int i = tid; i < kN * kH; i += 1024) { cacc[i] = 0.f; hacc[i] = 0.f; }
        for (int n = tid; n < kN; n += 1024) ipacc[n] = 0.f;

        // Preload WhT chunk for ht=0 while zeroing proceeds.
        issue_wchunk(WhT, 0, 0, wave, lane);
        wait_async0();
        __syncthreads();

        // ---- phase 1: z = Xz + h @ Wh -> LSTM gates -> c2,h2 ----
        // wave w owns node tile w; loops 16 hidden tiles with W double-buffer.
        {
            const int mt = wave;
            for (int ht = 0; ht < 16; ++ht) {
                const int cur = ht & 1;
                if (ht < 15) issue_wchunk(WhT, ht + 1, cur ^ 1, wave, lane);
                const __bf16* wch = wbuf + (size_t)cur * (LDS_WCHUNK_B / 2);

                v8f a0 = {0.f,0.f,0.f,0.f,0.f,0.f,0.f,0.f};
                v8f a1 = a0, a2 = a0, a3 = a0;
#pragma unroll
                for (int kt = 0; kt < 8; ++kt) {
                    v16bf afr = load_bf16_frag(hlds, mt * 16, kt * 32, kH);
                    v16bf b0 = load_bf16_frag(wch,  0 * 16,  kt * 32, kH);
                    v16bf b1 = load_bf16_frag(wch,  1 * 16,  kt * 32, kH);
                    v16bf b2 = load_bf16_frag(wch,  2 * 16,  kt * 32, kH);
                    v16bf b3 = load_bf16_frag(wch,  3 * 16,  kt * 32, kH);
                    a0 = WMMA_BF16(afr, b0, a0);   // gate i
                    a1 = WMMA_BF16(afr, b1, a1);   // gate f
                    a2 = WMMA_BF16(afr, b2, a2);   // gate g
                    a3 = WMMA_BF16(afr, b3, a3);   // gate o
                }
                const int hid = ht * 16 + lane15;
#pragma unroll
                for (int r = 0; r < 8; ++r) {
                    int node = mt * 16 + r + ((lane >> 4) << 3);
                    const float* xzrow = XzE + (size_t)node * (4 * kH);
                    float zi = a0[r] + xzrow[hid];
                    float zf = a1[r] + xzrow[kH + hid];
                    float zg = a2[r] + xzrow[2 * kH + hid];
                    float zo = a3[r] + xzrow[3 * kH + hid];
                    float cold = c[node * kH + hid];
                    float c2v = sigmoidf_(zf) * cold + sigmoidf_(zi) * tanhf(zg);
                    float h2v = sigmoidf_(zo) * tanhf(c2v);
                    c2b[node * kH + hid] = c2v;
                    h2b[node * kH + hid] = h2v;
                }
                wait_async0();
                __syncthreads();
            }
        }

        // ---- phase 2: branch softmax + probability-weighted scatter ----
        for (int q = 0; q < 16; ++q) {
            const int node = (q << 5) + wave;
            const bool isExit = (node == exitn);   // exit node keeps prior state
            const float* csrc = isExit ? (c + node * kH) : (c2b + node * kH);
            const float* hsrc = isExit ? (h + node * kH) : (h2b + node * kH);
            float d0 = 0.f, d1 = 0.f;
            for (int k = lane; k < kH; k += 32) {
                float xc = csrc[k];
                float xh = hsrc[k];
                d0 += xc * Wb[k * 2 + 0] + xh * Wb[(kH + k) * 2 + 0];
                d1 += xc * Wb[k * 2 + 1] + xh * Wb[(kH + k) * 2 + 1];
            }
#pragma unroll
            for (int off = 16; off >= 1; off >>= 1) {
                d0 += __shfl_xor(d0, off, 32);
                d1 += __shfl_xor(d1, off, 32);
            }
            d0 += bb[0]; d1 += bb[1];
            float m = fmaxf(d0, d1);
            float e0 = __expf(d0 - m), e1 = __expf(d1 - m);
            float inv = 1.0f / (e0 + e1);
            float ipn = ip[node];
            float pt = e0 * inv * ipn;
            float pf = e1 * inv * ipn;
            int t = tI[node], fi = fI[node];
            if (lane == 0) { atomicAdd(ipacc + t, pt); atomicAdd(ipacc + fi, pf); }
            for (int j = lane; j < kH; j += 32) {
                float cv = csrc[j], hv = hsrc[j];
                atomicAdd(cacc + t * kH + j, cv * pt);
                atomicAdd(cacc + fi * kH + j, cv * pf);
                atomicAdd(hacc + t * kH + j, hv * pt);
                atomicAdd(hacc + fi * kH + j, hv * pf);
            }
        }
        __syncthreads();

        // ---- phase 3: normalize by incoming probability, commit state ----
        for (int i = tid; i < kN * kH; i += 1024) {
            int n = i >> 8;
            float denom = ipacc[n] + 1e-7f;
            float cn = cacc[i] / denom;
            float hn = hacc[i] / denom;
            c[i] = cn;
            h[i] = hn;
            hlds[i] = (__bf16)hn;
        }
        for (int n = tid; n < kN; n += 1024) ip[n] = ipacc[n];
        __syncthreads();
    }
}

// ---------------------------------------------------------------------------
// Kernel 4: logits = concat(c[exit], h[exit]) @ Wo + bo
// Each block owns a 256-wide vocab slice and does all 16 examples, so Wo
// (64 MB) is streamed from HBM exactly once.
// ---------------------------------------------------------------------------
__global__ __launch_bounds__(256)
void ipagnn_out(const float* __restrict__ c, const float* __restrict__ h,
                const float* __restrict__ Wo, const float* __restrict__ bo,
                const int* __restrict__ exitIdx, float* __restrict__ out) {
    __shared__ float fin[kB][2 * kH];   // 32 KB
    for (int i = threadIdx.x; i < kB * 2 * kH; i += 256) {
        int b = i >> 9;
        int k = i & 511;
        int exitn = exitIdx[b];
        const float* row = (k < kH ? c : h) + ((size_t)(b * kN + exitn)) * kH;
        fin[b][k] = row[k & (kH - 1)];
    }
    __syncthreads();
    const int v = blockIdx.x * 256 + threadIdx.x;   // grid = kV/256 = 125
    float acc[kB];
    const float bias = bo[v];
#pragma unroll
    for (int b = 0; b < kB; ++b) acc[b] = bias;
    for (int k = 0; k < 2 * kH; ++k) {
        float w = Wo[(size_t)k * kV + v];
#pragma unroll
        for (int b = 0; b < kB; ++b) acc[b] += fin[b][k] * w;
    }
#pragma unroll
    for (int b = 0; b < kB; ++b) out[(size_t)b * kV + v] = acc[b];
}

// ---------------------------------------------------------------------------
extern "C" void kernel_launch(void* const* d_in, const int* in_sizes, int n_in,
                              void* d_out, int out_size, void* d_ws, size_t ws_size,
                              hipStream_t stream) {
    const float* node = (const float*)d_in[0];
    const float* Wx   = (const float*)d_in[1];
    const float* Wh   = (const float*)d_in[2];
    const float* b    = (const float*)d_in[3];
    const float* Wb   = (const float*)d_in[4];
    const float* bb   = (const float*)d_in[5];
    const float* Wo   = (const float*)d_in[6];
    const float* bo   = (const float*)d_in[7];
    const int* tIdx   = (const int*)d_in[8];
    const int* fIdx   = (const int*)d_in[9];
    const int* exitI  = (const int*)d_in[10];
    const int* steps  = (const int*)d_in[11];

    char* ws = (char*)d_ws;
    __bf16* WhT   = (__bf16*)(ws + OFF_WHT);
    __bf16* WxT   = (__bf16*)(ws + OFF_WXT);
    __bf16* nodeb = (__bf16*)(ws + OFF_NODEBF);
    float*  Xz    = (float*)(ws + OFF_XZ);
    float*  c     = (float*)(ws + OFF_C);
    float*  h     = (float*)(ws + OFF_H);
    float*  c2b   = (float*)(ws + OFF_C2);
    float*  h2b   = (float*)(ws + OFF_H2);
    float*  ip    = (float*)(ws + OFF_IP);
    float*  ipacc = (float*)(ws + OFF_IPACC);
    float*  cacc  = (float*)(ws + OFF_CACC);
    float*  hacc  = (float*)(ws + OFF_HACC);

    ipagnn_prep<<<2048, 256, 0, stream>>>(Wx, Wh, node, WxT, WhT, nodeb);
    ipagnn_xz<<<(512 * 64) / 8, 256, 0, stream>>>(nodeb, WxT, b, Xz);
    ipagnn_main<<<kB, 1024, LDS_TOTAL, stream>>>(Xz, WhT, c, h, c2b, h2b,
                                                 ip, ipacc, cacc, hacc,
                                                 Wb, bb, tIdx, fIdx, exitI, steps);
    ipagnn_out<<<kV / 256, 256, 0, stream>>>(c, h, Wo, bo, exitI, (float*)d_out);
}
